// SpaceFormerPretrain_1400159338536
// MI455X (gfx1250) — matrix operand order repaired and checked by reading, not compiled, not run on hardware
//
#include <hip/hip_runtime.h>
#include <math.h>

// ---------------------------------------------------------------------------
// SpaceFormer pretrain forward for MI455X (gfx1250, wave32, WMMA bf16)
// N=4096 cells, D=512, F=2048, n_mask=1024, n_zero=153
// ---------------------------------------------------------------------------

typedef __attribute__((ext_vector_type(16))) __bf16          bf16x16;
typedef __attribute__((ext_vector_type(8)))  __bf16          bf16x8;
typedef __attribute__((ext_vector_type(8)))  float           f32x8;
typedef __attribute__((ext_vector_type(8)))  unsigned short  u16x8;

union FragAB { bf16x16 v; bf16x8 h[2]; };

#define GAMMA_C 0.5f
#define NEG_C  -1e9f
#define LN_EPS_C 1e-5f

__device__ __forceinline__ unsigned short f32_to_bf16_rne(float f) {
    unsigned int u = __float_as_uint(f);
    unsigned int r = 0x7FFFu + ((u >> 16) & 1u);
    return (unsigned short)((u + r) >> 16);
}

__device__ __forceinline__ float gelu_exact(float x) {
    return 0.5f * x * (1.0f + erff(x * 0.7071067811865475f));
}

// ------------------------------- GEMM --------------------------------------
// C[M,Ncol] = A[M,K] @ Bt[Ncol,K]^T   (A, Bt bf16 row-major; acc f32)
// Block tile 128x128, 256 threads = 8 waves (2 along M x 4 along N),
// each wave computes 64x32 via 4x2 WMMA 16x16x32 bf16 fragments.
enum { MODE_BF16 = 0, MODE_SCORE = 1, MODE_GELU_BF16 = 2, MODE_BIAS_F32 = 3, MODE_F32 = 4 };

#define LDS_STRIDE 40   // 32 halfs + 8 pad (80B rows, 16B-aligned chunks)

__global__ __launch_bounds__(256)
void k_gemm(const unsigned short* __restrict__ A,
            const unsigned short* __restrict__ Bt,
            int M, int Ncol, int K,
            float* __restrict__ Cf,
            unsigned short* __restrict__ Cb,
            const unsigned char* __restrict__ maskp,
            const float* __restrict__ bias,
            float scale, int mode) {
    __shared__ unsigned short ldsA[128 * LDS_STRIDE];
    __shared__ unsigned short ldsB[128 * LDS_STRIDE];

    const int tid   = threadIdx.x;
    const int lane  = tid & 31;
    const int wave  = tid >> 5;          // 0..7
    const int wm    = wave >> 2;         // 0..1 -> M offset wm*64
    const int wn    = wave & 3;          // 0..3 -> N offset wn*32
    const int blkM  = blockIdx.y * 128;
    const int blkN  = blockIdx.x * 128;
    const int lm    = lane & 15;
    const int lhalf = lane >> 4;         // 0 or 1

    f32x8 acc[4][2] = {};

    const int aoff = lhalf * 8;          // A frag: K chunks {aoff..aoff+7, aoff+16..aoff+23}
    const int boff = lhalf * 8;          // B frag: K chunks {boff..boff+7, boff+16..boff+23}

    for (int k0 = 0; k0 < K; k0 += 32) {
#pragma unroll
        for (int it = 0; it < 2; ++it) {
            int c   = tid + it * 256;    // 512 16-byte chunks per tile
            int row = c >> 2;
            int sub = c & 3;
            *(u16x8*)&ldsA[row * LDS_STRIDE + sub * 8] =
                *(const u16x8*)(A + (size_t)(blkM + row) * K + k0 + sub * 8);
            *(u16x8*)&ldsB[row * LDS_STRIDE + sub * 8] =
                *(const u16x8*)(Bt + (size_t)(blkN + row) * K + k0 + sub * 8);
        }
        __syncthreads();

        FragAB af[4], bf[2];
#pragma unroll
        for (int f = 0; f < 4; ++f) {
            int r = wm * 64 + f * 16 + lm;
            af[f].h[0] = *(const bf16x8*)&ldsA[r * LDS_STRIDE + aoff];
            af[f].h[1] = *(const bf16x8*)&ldsA[r * LDS_STRIDE + aoff + 16];
        }
#pragma unroll
        for (int g = 0; g < 2; ++g) {
            int r = wn * 32 + g * 16 + lm;
            bf[g].h[0] = *(const bf16x8*)&ldsB[r * LDS_STRIDE + boff];
            bf[g].h[1] = *(const bf16x8*)&ldsB[r * LDS_STRIDE + boff + 16];
        }
#pragma unroll
        for (int f = 0; f < 4; ++f)
#pragma unroll
            for (int g = 0; g < 2; ++g)
                acc[f][g] = __builtin_amdgcn_wmma_f32_16x16x32_bf16(
                    false, af[f].v, false, bf[g].v, (short)0, acc[f][g], false, false);
        __syncthreads();
    }

    // Epilogue: C layout -> VGPR r, lanes 0-15: M=r, N=lane; lanes 16-31: M=r+8, N=lane-16
#pragma unroll
    for (int f = 0; f < 4; ++f) {
#pragma unroll
        for (int g = 0; g < 2; ++g) {
            int mBase = blkM + wm * 64 + f * 16 + lhalf * 8;
            int n     = blkN + wn * 32 + g * 16 + lm;
#pragma unroll
            for (int r = 0; r < 8; ++r) {
                int m = mBase + r;
                float v = acc[f][g][r];
                size_t idx = (size_t)m * Ncol + n;
                if (mode == MODE_BF16) {
                    Cb[idx] = f32_to_bf16_rne(v);
                } else if (mode == MODE_SCORE) {
                    Cf[idx] = maskp[idx] ? NEG_C : v * scale;
                } else if (mode == MODE_GELU_BF16) {
                    Cb[idx] = f32_to_bf16_rne(gelu_exact(v + bias[n]));
                } else if (mode == MODE_BIAS_F32) {
                    Cf[idx] = v + bias[n];
                } else {
                    Cf[idx] = v;
                }
            }
        }
    }
}

// --------------------------- helper kernels --------------------------------

__global__ __launch_bounds__(256)
void k_copy_convert(const float* __restrict__ x, float* __restrict__ of,
                    unsigned short* __restrict__ ob, int n) {
    int i = blockIdx.x * 256 + threadIdx.x;
    if (i < n) { float v = x[i]; of[i] = v; ob[i] = f32_to_bf16_rne(v); }
}

__global__ __launch_bounds__(256)
void k_scatter_zero(float* __restrict__ uf, unsigned short* __restrict__ ub,
                    const int* __restrict__ mask_nodes, const int* __restrict__ gene_idx,
                    int nm, int nz, int D) {
    int i = blockIdx.x * 256 + threadIdx.x;
    if (i < nm * nz) {
        int row = mask_nodes[i / nz];
        int col = gene_idx[i];
        size_t idx = (size_t)row * D + col;
        uf[idx] = 0.0f;
        ub[idx] = 0;
    }
}

__global__ __launch_bounds__(256)
void k_trans_f2b(const float* __restrict__ in, unsigned short* __restrict__ out,
                 int R, int C) {
    int i = blockIdx.x * 256 + threadIdx.x;
    if (i < R * C) {
        int r = i / C, c = i % C;
        out[(size_t)c * R + r] = f32_to_bf16_rne(in[i]);
    }
}

__global__ __launch_bounds__(256)
void k_trans_b2b(const unsigned short* __restrict__ in, unsigned short* __restrict__ out,
                 int R, int C) {
    int i = blockIdx.x * 256 + threadIdx.x;
    if (i < R * C) {
        int r = i / C, c = i % C;
        out[(size_t)c * R + r] = in[i];
    }
}

__global__ __launch_bounds__(256)
void k_rowmax(const float* __restrict__ sr, const float* __restrict__ sf,
              float* __restrict__ out, int Ncol) {
    __shared__ float red[256];
    int row = blockIdx.x, t = threadIdx.x;
    size_t base = (size_t)row * Ncol;
    float m = -3.4e38f;
    for (int j = t; j < Ncol; j += 256) {
        m = fmaxf(m, sr[base + j]);
        m = fmaxf(m, sf[base + j]);
    }
    red[t] = m; __syncthreads();
    for (int s = 128; s > 0; s >>= 1) {
        if (t < s) red[t] = fmaxf(red[t], red[t + s]);
        __syncthreads();
    }
    if (t == 0) out[row] = red[0];
}

__global__ __launch_bounds__(256)
void k_combine(float* __restrict__ sr, const float* __restrict__ sf,
               const float* __restrict__ rowmax,
               float* __restrict__ attnF,            // nullable
               unsigned short* __restrict__ attnB, int Ncol) {
    __shared__ float red[256];
    __shared__ float invTot;
    int row = blockIdx.x, t = threadIdx.x;
    size_t base = (size_t)row * Ncol;
    float mx = rowmax[row];
    float s = 0.0f;
    for (int j = t; j < Ncol; j += 256) {
        float e = expf(sr[base + j] - mx) + GAMMA_C * expf(sf[base + j] - mx);
        sr[base + j] = e;
        s += e;
    }
    red[t] = s; __syncthreads();
    for (int st = 128; st > 0; st >>= 1) {
        if (t < st) red[t] += red[t + st];
        __syncthreads();
    }
    if (t == 0) invTot = 1.0f / red[0];
    __syncthreads();
    float inv = invTot;
    for (int j = t; j < Ncol; j += 256) {
        float a = sr[base + j] * inv;
        if (attnF) attnF[base + j] = a;
        attnB[base + j] = f32_to_bf16_rne(a);
    }
}

// out = LayerNorm(a + b) * g + beta ; writes f32 + bf16.  One row (D=512) per block.
__global__ __launch_bounds__(256)
void k_add_ln(const float* __restrict__ a, const float* __restrict__ b,
              const float* __restrict__ g, const float* __restrict__ be,
              float* __restrict__ outF, unsigned short* __restrict__ outB, int D) {
    __shared__ float red[256];
    __shared__ float stats[2];
    int row = blockIdx.x, t = threadIdx.x;
    size_t base = (size_t)row * D;
    float v0 = a[base + t] + b[base + t];
    float v1 = a[base + t + 256] + b[base + t + 256];
    red[t] = v0 + v1; __syncthreads();
    for (int s = 128; s > 0; s >>= 1) {
        if (t < s) red[t] += red[t + s];
        __syncthreads();
    }
    if (t == 0) stats[0] = red[0] / (float)D;
    __syncthreads();
    float mean = stats[0];
    float d0 = v0 - mean, d1 = v1 - mean;
    red[t] = d0 * d0 + d1 * d1; __syncthreads();
    for (int s = 128; s > 0; s >>= 1) {
        if (t < s) red[t] += red[t + s];
        __syncthreads();
    }
    if (t == 0) stats[1] = rsqrtf(red[0] / (float)D + LN_EPS_C);
    __syncthreads();
    float inv = stats[1];
    float y0 = d0 * inv * g[t] + be[t];
    float y1 = d1 * inv * g[t + 256] + be[t + 256];
    outF[base + t] = y0;       outF[base + t + 256] = y1;
    outB[base + t] = f32_to_bf16_rne(y0);
    outB[base + t + 256] = f32_to_bf16_rne(y1);
}

__global__ __launch_bounds__(256)
void k_gather(const float* __restrict__ x, const float* __restrict__ h,
              const int* __restrict__ mask_nodes, const int* __restrict__ gene_idx,
              float* __restrict__ xinit, float* __restrict__ xrecon,
              int nm, int nz, int D) {
    int i = blockIdx.x * 256 + threadIdx.x;
    if (i < nm * nz) {
        int row = mask_nodes[i / nz];
        int col = gene_idx[i];
        size_t idx = (size_t)row * D + col;
        xinit[i]  = x[idx];
        xrecon[i] = h[idx];
    }
}

// ------------------------------- driver ------------------------------------

extern "C" void kernel_launch(void* const* d_in, const int* in_sizes, int n_in,
                              void* d_out, int out_size, void* d_ws, size_t ws_size,
                              hipStream_t stream) {
    const int Nn = 4096, Dd = 512, Ff = 2048, NM = 1024, NZ = 153;
    const float scale = 0.04419417382415922f;   // 1/sqrt(512)

    // ---- inputs (setup_inputs dict order, depth-first) ----
    int q = 0;
    const float* x = (const float*)d_in[q++];
    const float* encW[5]; for (int i = 0; i < 5; ++i) encW[i] = (const float*)d_in[q++];
    const float* decW[5]; for (int i = 0; i < 5; ++i) decW[i] = (const float*)d_in[q++];
    const float* ln[8];   for (int i = 0; i < 8; ++i) ln[i]   = (const float*)d_in[q++];
    const float* ff_w1  = (const float*)d_in[q++]; const float* ff_b1  = (const float*)d_in[q++];
    const float* ff_w2  = (const float*)d_in[q++]; const float* ff_b2  = (const float*)d_in[q++];
    const float* ff1_w1 = (const float*)d_in[q++]; const float* ff1_b1 = (const float*)d_in[q++];
    const float* ff1_w2 = (const float*)d_in[q++]; const float* ff1_b2 = (const float*)d_in[q++];
    const float* head_w = (const float*)d_in[q++]; const float* head_b = (const float*)d_in[q++];
    const unsigned char* realm = (const unsigned char*)d_in[q++];
    const unsigned char* fakem = (const unsigned char*)d_in[q++];
    const int* mask_nodes = (const int*)d_in[q++];
    const int* gene_idx   = (const int*)d_in[q++];

    // ---- outputs ----
    float* out_xinit  = (float*)d_out;
    float* out_xrecon = out_xinit + (size_t)NM * NZ;
    float* out_attn   = out_xrecon + (size_t)NM * NZ;
    float* out_recon  = out_attn + (size_t)Nn * Nn;

    // ---- workspace carve ----
    char* wptr = (char*)d_ws;
    auto carve = [&](size_t bytes) -> void* {
        void* p = (void*)wptr;
        wptr += (bytes + 255) & ~(size_t)255;
        return p;
    };
    float* use_x_f = (float*)carve((size_t)Nn * Dd * 4);
    float* hid_f   = (float*)carve((size_t)Nn * Dd * 4);
    float* tmp_f   = (float*)carve((size_t)Nn * Dd * 4);
    unsigned short* use_x_b = (unsigned short*)carve((size_t)Nn * Dd * 2);
    unsigned short* hid_b   = (unsigned short*)carve((size_t)Nn * Dd * 2);
    unsigned short* recon_b = (unsigned short*)carve((size_t)Nn * Dd * 2);
    unsigned short* Qr = (unsigned short*)carve((size_t)Nn * Dd * 2);
    unsigned short* Kr = (unsigned short*)carve((size_t)Nn * Dd * 2);
    unsigned short* Qf = (unsigned short*)carve((size_t)Nn * Dd * 2);
    unsigned short* Kf = (unsigned short*)carve((size_t)Nn * Dd * 2);
    unsigned short* Vb = (unsigned short*)carve((size_t)Nn * Dd * 2);
    unsigned short* VT = (unsigned short*)carve((size_t)Nn * Dd * 2);
    unsigned short* ffmid_b = (unsigned short*)carve((size_t)Nn * Ff * 2);
    unsigned short* encT[5], *decT[5];
    for (int i = 0; i < 5; ++i) encT[i] = (unsigned short*)carve((size_t)Dd * Dd * 2);
    for (int i = 0; i < 5; ++i) decT[i] = (unsigned short*)carve((size_t)Dd * Dd * 2);
    unsigned short* ffw1T  = (unsigned short*)carve((size_t)Dd * Ff * 2);
    unsigned short* ffw2T  = (unsigned short*)carve((size_t)Ff * Dd * 2);
    unsigned short* ff1w1T = (unsigned short*)carve((size_t)Dd * Ff * 2);
    unsigned short* ff1w2T = (unsigned short*)carve((size_t)Ff * Dd * 2);
    unsigned short* headT  = (unsigned short*)carve((size_t)Dd * Dd * 2);
    float* sr = (float*)carve((size_t)Nn * Nn * 4);
    float* sf = (float*)carve((size_t)Nn * Nn * 4);
    unsigned short* attn_b = (unsigned short*)carve((size_t)Nn * Nn * 2);
    float* rowmax_buf = (float*)carve((size_t)Nn * 4);

    auto gemm = [&](const unsigned short* A, const unsigned short* Bt, int M, int Ncol, int K,
                    float* Cf, unsigned short* Cb, const unsigned char* mk, const float* bias,
                    float sc, int mode) {
        dim3 g(Ncol / 128, M / 128);
        k_gemm<<<g, 256, 0, stream>>>(A, Bt, M, Ncol, K, Cf, Cb, mk, bias, sc, mode);
    };
    auto nblk = [](int n) { return (n + 255) / 256; };

    // ---- 1. use_x = x with masked genes zeroed (f32 + bf16) ----
    k_copy_convert<<<nblk(Nn * Dd), 256, 0, stream>>>(x, use_x_f, use_x_b, Nn * Dd);
    k_scatter_zero<<<nblk(NM * NZ), 256, 0, stream>>>(use_x_f, use_x_b, mask_nodes, gene_idx,
                                                      NM, NZ, Dd);

    // ---- 2. transpose+convert all weights to bf16 [N,K] ----
    for (int i = 0; i < 5; ++i) {
        k_trans_f2b<<<nblk(Dd * Dd), 256, 0, stream>>>(encW[i], encT[i], Dd, Dd);
        k_trans_f2b<<<nblk(Dd * Dd), 256, 0, stream>>>(decW[i], decT[i], Dd, Dd);
    }
    k_trans_f2b<<<nblk(Dd * Ff), 256, 0, stream>>>(ff_w1,  ffw1T,  Dd, Ff);
    k_trans_f2b<<<nblk(Ff * Dd), 256, 0, stream>>>(ff_w2,  ffw2T,  Ff, Dd);
    k_trans_f2b<<<nblk(Dd * Ff), 256, 0, stream>>>(ff1_w1, ff1w1T, Dd, Ff);
    k_trans_f2b<<<nblk(Ff * Dd), 256, 0, stream>>>(ff1_w2, ff1w2T, Ff, Dd);
    k_trans_f2b<<<nblk(Dd * Dd), 256, 0, stream>>>(head_w, headT,  Dd, Dd);

    // ---- spatial attention block (shared by enc/dec) -> tmp_f = attn @ V ----
    auto attn_block = [&](const unsigned short* inB, unsigned short* const* WT, float* attnF) {
        gemm(inB, WT[0], Nn, Dd, Dd, nullptr, Qr, nullptr, nullptr, 0.f, MODE_BF16);
        gemm(inB, WT[1], Nn, Dd, Dd, nullptr, Kr, nullptr, nullptr, 0.f, MODE_BF16);
        gemm(inB, WT[2], Nn, Dd, Dd, nullptr, Qf, nullptr, nullptr, 0.f, MODE_BF16);
        gemm(inB, WT[3], Nn, Dd, Dd, nullptr, Kf, nullptr, nullptr, 0.f, MODE_BF16);
        gemm(inB, WT[4], Nn, Dd, Dd, nullptr, Vb, nullptr, nullptr, 0.f, MODE_BF16);
        k_trans_b2b<<<nblk(Nn * Dd), 256, 0, stream>>>(Vb, VT, Nn, Dd);
        gemm(Qr, Kr, Nn, Nn, Dd, sr, nullptr, realm, nullptr, scale, MODE_SCORE);
        gemm(Qf, Kf, Nn, Nn, Dd, sf, nullptr, fakem, nullptr, scale, MODE_SCORE);
        k_rowmax<<<Nn, 256, 0, stream>>>(sr, sf, rowmax_buf, Nn);
        k_combine<<<Nn, 256, 0, stream>>>(sr, sf, rowmax_buf, attnF, attn_b, Nn);
        gemm(attn_b, VT, Nn, Dd, Nn, tmp_f, nullptr, nullptr, nullptr, 0.f, MODE_F32);
    };

    // ---- 3. encoder ----
    attn_block(use_x_b, encT, out_attn);                                  // encode_weights out
    k_add_ln<<<Nn, 256, 0, stream>>>(tmp_f, use_x_f, ln[0], ln[1], hid_f, hid_b, Dd);
    gemm(hid_b, ffw1T, Nn, Ff, Dd, nullptr, ffmid_b, nullptr, ff_b1, 0.f, MODE_GELU_BF16);
    gemm(ffmid_b, ffw2T, Nn, Dd, Ff, tmp_f, nullptr, nullptr, ff_b2, 0.f, MODE_BIAS_F32);
    k_add_ln<<<Nn, 256, 0, stream>>>(hid_f, tmp_f, ln[2], ln[3], hid_f, hid_b, Dd);

    // ---- 4. decoder ----
    attn_block(hid_b, decT, nullptr);
    k_add_ln<<<Nn, 256, 0, stream>>>(tmp_f, hid_f, ln[4], ln[5], hid_f, hid_b, Dd);
    gemm(hid_b, ff1w1T, Nn, Ff, Dd, nullptr, ffmid_b, nullptr, ff1_b1, 0.f, MODE_GELU_BF16);
    gemm(ffmid_b, ff1w2T, Nn, Dd, Ff, tmp_f, nullptr, nullptr, ff1_b2, 0.f, MODE_BIAS_F32);
    k_add_ln<<<Nn, 256, 0, stream>>>(hid_f, tmp_f, ln[6], ln[7], out_recon, recon_b, Dd);  // recon out

    // ---- 5. head + gathers ----
    gemm(recon_b, headT, Nn, Dd, Dd, tmp_f, nullptr, nullptr, head_b, 0.f, MODE_BIAS_F32);
    k_gather<<<nblk(NM * NZ), 256, 0, stream>>>(x, tmp_f, mask_nodes, gene_idx,
                                                out_xinit, out_xrecon, NM, NZ, Dd);

    (void)in_sizes; (void)n_in; (void)out_size; (void)ws_size;
}
